// RWKV_Model_24764781429094
// MI455X (gfx1250) — matrix operand order repaired and checked by reading, not compiled, not run on hardware
//
#include <hip/hip_runtime.h>
#include <cstdint>
#include <cstddef>

// ---------------- model dims ----------------
constexpr int DD = 128;   // model dim
constexpr int FF = 512;   // ffn dim
constexpr int VV = 256;   // vocab
constexpr int LL = 2;     // layers
constexpr int TT = 1024;  // seq len
constexpr int BB = 32;    // batch

typedef __bf16 v16bf __attribute__((ext_vector_type(16)));
typedef __bf16 v8bf  __attribute__((ext_vector_type(8)));
typedef float  v8f   __attribute__((ext_vector_type(8)));

// fp32 -> bf16 bits (round to nearest even)
__device__ __forceinline__ unsigned short f2bf(float f) {
  uint32_t u = __builtin_bit_cast(uint32_t, f);
  uint32_t r = u + 0x7FFFu + ((u >> 16) & 1u);
  return (unsigned short)(r >> 16);
}

// ---- WMMA fragment loads (contiguous k order, same for A and B, wave32) ----
// A 16x32 (MxK): lane m=lane&15; lanes0-15 hold K 0..15, lanes16-31 hold K 16..31
__device__ __forceinline__ v16bf load_a_frag(const unsigned short* base, int stride,
                                             int kbase, int lane) {
  const int m  = lane & 15;
  const int ko = (lane >> 4) << 4;
  const unsigned short* p = base + m * stride + kbase + ko;
  union { v16bf v; v8bf h[2]; } u;
  u.h[0] = *reinterpret_cast<const v8bf*>(p);
  u.h[1] = *reinterpret_cast<const v8bf*>(p + 8);
  return u.v;
}

// B 32x16 (KxN): lane col n=lane&15; lanes0-15 hold K 0..15, lanes16-31 hold K 16..31.
// Weights stored row-major [out][in] => 16 contiguous K per lane.
__device__ __forceinline__ v16bf load_b_frag(const unsigned short* W, int kstride,
                                             int nbase, int kbase, int lane) {
  const int n  = nbase + (lane & 15);
  const int kk = kbase + ((lane >> 4) << 4);
  return *reinterpret_cast<const v16bf*>(W + (size_t)n * kstride + kk);
}

// Out[16][nbase..+16) += A(16xK, bf16 LDS) * W(KxN, bf16 row-major [n][k])
__device__ __forceinline__ v8f mm16(const unsigned short* A, int astride,
                                    const unsigned short* W, int K,
                                    int nbase, int lane, v8f acc) {
  for (int kb = 0; kb < K; kb += 32) {
    v16bf a = load_a_frag(A, astride, kb, lane);
    v16bf b = load_b_frag(W, K, nbase, kb, lane);
    acc = __builtin_amdgcn_wmma_f32_16x16x32_bf16(false, a, false, b,
                                                  (short)0, acc, false, false);
  }
  return acc;
}

__device__ __forceinline__ float wsum32(float v) {
  for (int o = 16; o > 0; o >>= 1) v += __shfl_xor(v, o, 32);
  return v;
}

// ---------------- prep kernels ----------------
__global__ void cvt_bf16(const float* __restrict__ src, unsigned short* __restrict__ dst, int n) {
  int i = blockIdx.x * blockDim.x + threadIdx.x;
  if (i < n) dst[i] = f2bf(src[i]);
}

__global__ void init_state(float* st, int n, int spLo, int spHi) {
  int i = blockIdx.x * blockDim.x + threadIdx.x;
  if (i < n) st[i] = (i >= spLo && i < spHi) ? -1e30f : 0.0f;
}

// ---------------- main persistent kernel: one block = 16 batch rows, whole T loop ----------------
__global__ __launch_bounds__(256)
void rwkv_main(const int* __restrict__ tokens, const float* __restrict__ emb,
               const float* __restrict__ ln0_g, const float* __restrict__ ln0_b,
               const float* __restrict__ ln1_g, const float* __restrict__ ln1_b,
               const float* __restrict__ ln2_g, const float* __restrict__ ln2_b,
               const float* __restrict__ td,   const float* __restrict__ tf,
               const float* __restrict__ tmk,  const float* __restrict__ tmv,
               const float* __restrict__ tmr,
               const float* __restrict__ cmk,  const float* __restrict__ cmr,
               const float* __restrict__ lnout_g, const float* __restrict__ lnout_b,
               const unsigned short* __restrict__ Wkb, const unsigned short* __restrict__ Wvb,
               const unsigned short* __restrict__ Wrb, const unsigned short* __restrict__ Wob,
               const unsigned short* __restrict__ Ckb, const unsigned short* __restrict__ Cvb,
               const unsigned short* __restrict__ Crb, const unsigned short* __restrict__ headb,
               float* __restrict__ sx, float* __restrict__ sA, float* __restrict__ sB,
               float* __restrict__ sp, float* __restrict__ sf,
               float* __restrict__ out) {
  // LDS arena (48 KB)
  __shared__ __align__(32) unsigned char lds[49152];
  float* xbuf = (float*)(lds + 0);             // [16][128] running x
  float* t0   = (float*)(lds + 8192);          // [16][128] k
  float* t1   = (float*)(lds + 16384);         // [16][128] v
  float* t2   = (float*)(lds + 24576);         // [16][128] r / gate
  unsigned short* abuf = (unsigned short*)(lds + 32768); // [16][512] bf16 A operands
  unsigned short* kfb  = (unsigned short*)(lds + 8192);  // [16][512] bf16 kf (aliases t0,t1)

  const int tid  = threadIdx.x;
  const int wave = tid >> 5;
  const int lane = tid & 31;
  const int b0   = blockIdx.x * 16;     // batch tile base
  const int hi   = lane >> 4, col = lane & 15;

  for (int t = 0; t < TT; ++t) {
    // ---- embedding + LN0 -> xbuf ----
    for (int rr = 0; rr < 2; ++rr) {
      const int m = wave * 2 + rr;
      const int b = b0 + m;
      const int tok = tokens[b * TT + t];
      float v[4]; float s = 0.f;
      for (int i = 0; i < 4; ++i) { int d = lane + 32 * i; v[i] = emb[tok * DD + d]; s += v[i]; }
      s = wsum32(s) * (1.0f / DD);
      float q = 0.f;
      for (int i = 0; i < 4; ++i) { float c = v[i] - s; q += c * c; }
      q = wsum32(q) * (1.0f / DD);
      const float inv = rsqrtf(q + 1e-5f);
      for (int i = 0; i < 4; ++i) {
        int d = lane + 32 * i;
        xbuf[m * DD + d] = (v[i] - s) * inv * ln0_g[d] + ln0_b[d];
      }
    }
    __syncthreads();

    for (int l = 0; l < LL; ++l) {
      const int lo = l * DD;
      // ---- phase 1: LN1, time-mix interpolation, state sx update ----
      for (int rr = 0; rr < 2; ++rr) {
        const int m = wave * 2 + rr;
        const int b = b0 + m;
        const size_t sbase = ((size_t)l * BB + b) * DD;
        float v[4]; float s = 0.f;
        for (int i = 0; i < 4; ++i) { int d = lane + 32 * i; v[i] = xbuf[m * DD + d]; s += v[i]; }
        s = wsum32(s) * (1.0f / DD);
        float q = 0.f;
        for (int i = 0; i < 4; ++i) { float c = v[i] - s; q += c * c; }
        q = wsum32(q) * (1.0f / DD);
        const float inv = rsqrtf(q + 1e-5f);
        for (int i = 0; i < 4; ++i) {
          const int d = lane + 32 * i;
          const float h = (v[i] - s) * inv * ln1_g[lo + d] + ln1_b[lo + d];
          const float so = sx[sbase + d];
          float mk = tmk[lo + d], mv = tmv[lo + d], mr = tmr[lo + d];
          abuf[m * 512 +       d] = f2bf(h * mk + so * (1.f - mk));
          abuf[m * 512 + 128 + d] = f2bf(h * mv + so * (1.f - mv));
          abuf[m * 512 + 256 + d] = f2bf(h * mr + so * (1.f - mr));
          sx[sbase + d] = h;
        }
      }
      __syncthreads();
      // ---- phase 2: k, v, r GEMMs (wave owns N-tile) ----
      {
        const int nbase = wave * 16;
        v8f ak = {}; ak = mm16(abuf,       512, Wkb + l * DD * DD, DD, nbase, lane, ak);
        v8f av = {}; av = mm16(abuf + 128, 512, Wvb + l * DD * DD, DD, nbase, lane, av);
        v8f ar = {}; ar = mm16(abuf + 256, 512, Wrb + l * DD * DD, DD, nbase, lane, ar);
        for (int r = 0; r < 8; ++r) {
          const int m = r + 8 * hi;
          t0[m * DD + nbase + col] = ak[r];
          t1[m * DD + nbase + col] = av[r];
          t2[m * DD + nbase + col] = 1.0f / (1.0f + __expf(-ar[r]));
        }
      }
      __syncthreads();
      // ---- phase 3: WKV recurrence (elementwise, state in L2) ----
      for (int i = tid; i < 16 * DD; i += 256) {
        const int m = i >> 7, d = i & 127;
        const int b = b0 + m;
        const size_t si = ((size_t)l * BB + b) * DD + d;
        const float k = t0[i], vv = t1[i], r = t2[i];
        const float spv = sp[si], sAv = sA[si], sBv = sB[si];
        const float ww = tf[lo + d] + k;
        const float p  = fmaxf(spv, ww);
        float e1 = __expf(spv - p), e2 = __expf(ww - p);
        const float wkv = (e1 * sAv + e2 * vv) / (e1 * sBv + e2 + 1e-8f);
        abuf[m * 512 + 384 + d] = f2bf(r * wkv);
        const float ww2 = spv + td[lo + d];
        const float p2  = fmaxf(ww2, k);
        e1 = __expf(ww2 - p2); e2 = __expf(k - p2);
        sA[si] = e1 * sAv + e2 * vv;
        sB[si] = e1 * sBv + e2;
        sp[si] = p2;
      }
      __syncthreads();
      // ---- phase 4: output projection, x += (r*wkv) @ Wo^T ----
      {
        const int nbase = wave * 16;
        v8f ao = {}; ao = mm16(abuf + 384, 512, Wob + l * DD * DD, DD, nbase, lane, ao);
        for (int r = 0; r < 8; ++r) {
          const int m = r + 8 * hi;
          xbuf[m * DD + nbase + col] += ao[r];
        }
      }
      __syncthreads();
      // ---- phase 5: LN2, channel-mix interpolation, state sf update ----
      for (int rr = 0; rr < 2; ++rr) {
        const int m = wave * 2 + rr;
        const int b = b0 + m;
        const size_t sbase = ((size_t)l * BB + b) * DD;
        float v[4]; float s = 0.f;
        for (int i = 0; i < 4; ++i) { int d = lane + 32 * i; v[i] = xbuf[m * DD + d]; s += v[i]; }
        s = wsum32(s) * (1.0f / DD);
        float q = 0.f;
        for (int i = 0; i < 4; ++i) { float c = v[i] - s; q += c * c; }
        q = wsum32(q) * (1.0f / DD);
        const float inv = rsqrtf(q + 1e-5f);
        for (int i = 0; i < 4; ++i) {
          const int d = lane + 32 * i;
          const float h2 = (v[i] - s) * inv * ln2_g[lo + d] + ln2_b[lo + d];
          const float so = sf[sbase + d];
          float mk = cmk[lo + d], mr = cmr[lo + d];
          abuf[m * 512 +       d] = f2bf(h2 * mk + so * (1.f - mk));
          abuf[m * 512 + 128 + d] = f2bf(h2 * mr + so * (1.f - mr));
          sf[sbase + d] = h2;
        }
      }
      __syncthreads();
      // ---- phase 6a: kf = relu(xk2 @ Ck^T)^2 -> bf16 (F=512, 4 tiles/wave); gate = sigmoid(xr2 @ Cr^T) ----
      for (int j = 0; j < 4; ++j) {
        const int nbase = (wave + 8 * j) * 16;
        v8f ac = {}; ac = mm16(abuf, 512, Ckb + (size_t)l * FF * DD, DD, nbase, lane, ac);
        for (int r = 0; r < 8; ++r) {
          const int m = r + 8 * hi;
          float x = ac[r]; x = x > 0.f ? x * x : 0.f;
          kfb[m * FF + nbase + col] = f2bf(x);
        }
      }
      {
        const int nbase = wave * 16;
        v8f ag = {}; ag = mm16(abuf + 128, 512, Crb + l * DD * DD, DD, nbase, lane, ag);
        for (int r = 0; r < 8; ++r) {
          const int m = r + 8 * hi;
          t2[m * DD + nbase + col] = 1.0f / (1.0f + __expf(-ag[r]));
        }
      }
      __syncthreads();
      // ---- phase 6b: x += gate * (kf @ Cv^T)  (K=512) ----
      {
        const int nbase = wave * 16;
        v8f acv = {}; acv = mm16(kfb, FF, Cvb + (size_t)l * DD * FF, FF, nbase, lane, acv);
        for (int r = 0; r < 8; ++r) {
          const int m = r + 8 * hi;
          xbuf[m * DD + nbase + col] += t2[m * DD + nbase + col] * acv[r];
        }
      }
      __syncthreads();
    } // layers

    // ---- final LN -> abuf, head GEMM -> logits ----
    for (int rr = 0; rr < 2; ++rr) {
      const int m = wave * 2 + rr;
      float v[4]; float s = 0.f;
      for (int i = 0; i < 4; ++i) { int d = lane + 32 * i; v[i] = xbuf[m * DD + d]; s += v[i]; }
      s = wsum32(s) * (1.0f / DD);
      float q = 0.f;
      for (int i = 0; i < 4; ++i) { float c = v[i] - s; q += c * c; }
      q = wsum32(q) * (1.0f / DD);
      const float inv = rsqrtf(q + 1e-5f);
      for (int i = 0; i < 4; ++i) {
        const int d = lane + 32 * i;
        abuf[m * 512 + d] = f2bf((v[i] - s) * inv * lnout_g[d] + lnout_b[d]);
      }
    }
    __syncthreads();
    for (int j = 0; j < 2; ++j) {
      const int nbase = (wave + 8 * j) * 16;
      v8f ah = {}; ah = mm16(abuf, 512, headb, DD, nbase, lane, ah);
      for (int r = 0; r < 8; ++r) {
        const int m = r + 8 * hi;
        const int b = b0 + m;
        out[((size_t)b * TT + t) * VV + nbase + col] = ah[r];
      }
    }
    __syncthreads();
  } // t
}

// ---------------- host launcher ----------------
extern "C" void kernel_launch(void* const* d_in, const int* in_sizes, int n_in,
                              void* d_out, int out_size, void* d_ws, size_t ws_size,
                              hipStream_t stream) {
  (void)in_sizes; (void)n_in; (void)out_size; (void)ws_size;
  const int*   tokens  = (const int*)d_in[0];
  const float* emb     = (const float*)d_in[1];
  const float* ln0_g   = (const float*)d_in[2];
  const float* ln0_b   = (const float*)d_in[3];
  const float* ln1_g   = (const float*)d_in[4];
  const float* ln1_b   = (const float*)d_in[5];
  const float* ln2_g   = (const float*)d_in[6];
  const float* ln2_b   = (const float*)d_in[7];
  const float* td      = (const float*)d_in[8];
  const float* tf      = (const float*)d_in[9];
  const float* tmk     = (const float*)d_in[10];
  const float* tmv     = (const float*)d_in[11];
  const float* tmr     = (const float*)d_in[12];
  const float* Wk      = (const float*)d_in[13];
  const float* Wv      = (const float*)d_in[14];
  const float* Wr      = (const float*)d_in[15];
  const float* Wo      = (const float*)d_in[16];
  const float* cmk     = (const float*)d_in[17];
  const float* cmr     = (const float*)d_in[18];
  const float* Ck      = (const float*)d_in[19];
  const float* Cv      = (const float*)d_in[20];
  const float* Cr      = (const float*)d_in[21];
  const float* lnout_g = (const float*)d_in[22];
  const float* lnout_b = (const float*)d_in[23];
  const float* head    = (const float*)d_in[24];

  // workspace layout: bf16 weights then fp32 state
  const int nW  = LL * DD * DD;   // 32768
  const int nC  = LL * FF * DD;   // 131072
  const int nH  = VV * DD;        // 32768
  unsigned short* wsb   = (unsigned short*)d_ws;
  unsigned short* Wkb   = wsb;
  unsigned short* Wvb   = Wkb + nW;
  unsigned short* Wrb   = Wvb + nW;
  unsigned short* Wob   = Wrb + nW;
  unsigned short* Ckb   = Wob + nW;
  unsigned short* Cvb   = Ckb + nC;
  unsigned short* Crb   = Cvb + nC;
  unsigned short* headb = Crb + nW;
  const size_t wbytes = (size_t)(4 * nW + 2 * nC + nW + nH) * sizeof(unsigned short);

  float* st = (float*)((char*)d_ws + wbytes);
  const int nS = LL * BB * DD;    // 8192 per state array
  float* sx = st;
  float* sA = st + 1 * nS;
  float* sB = st + 2 * nS;
  float* sp = st + 3 * nS;
  float* sf = st + 4 * nS;

  auto cvt = [&](const float* s, unsigned short* d, int n) {
    cvt_bf16<<<(n + 255) / 256, 256, 0, stream>>>(s, d, n);
  };
  cvt(Wk, Wkb, nW); cvt(Wv, Wvb, nW); cvt(Wr, Wrb, nW); cvt(Wo, Wob, nW);
  cvt(Ck, Ckb, nC); cvt(Cv, Cvb, nC); cvt(Cr, Crb, nW); cvt(head, headb, nH);

  const int nState = 5 * nS;
  init_state<<<(nState + 255) / 256, 256, 0, stream>>>(st, nState, 3 * nS, 4 * nS);

  rwkv_main<<<dim3(BB / 16), dim3(256), 0, stream>>>(
      tokens, emb, ln0_g, ln0_b, ln1_g, ln1_b, ln2_g, ln2_b, td, tf,
      tmk, tmv, tmr, cmk, cmr, lnout_g, lnout_b,
      Wkb, Wvb, Wrb, Wob, Ckb, Cvb, Crb, headb,
      sx, sA, sB, sp, sf, (float*)d_out);
}
